// MLPBaseline_75617194213652
// MI455X (gfx1250) — matrix operand order repaired and checked by reading, compiled, not run on hardware
//
#include <hip/hip_runtime.h>
#include <hip/hip_bf16.h>

typedef __bf16        bf16x16 __attribute__((ext_vector_type(16)));
typedef float         f32x8   __attribute__((ext_vector_type(8)));
typedef unsigned int  u32x8   __attribute__((ext_vector_type(8)));

#define D_NODE 64
#define D_EDGE 32
#define D_IN   160
#define H1     128
#define H2     64
#define NC     8
#define WAVES_PER_BLOCK 4
#define TPB (WAVES_PER_BLOCK * 32)

// ---- f32 -> bf16 packing ----
// RNE-ish (round-half-up) pack: used once for weights in the prologue.
#if __has_builtin(__builtin_amdgcn_cvt_pk_bf16_f32)
__device__ __forceinline__ unsigned int pkbf_rnd(float lo, float hi) {
  return __builtin_bit_cast(unsigned int, __builtin_amdgcn_cvt_pk_bf16_f32(lo, hi));
}
#else
__device__ __forceinline__ unsigned int pkbf_rnd(float lo, float hi) {
  unsigned int a = __float_as_uint(lo) + 0x8000u;
  unsigned int b = __float_as_uint(hi) + 0x8000u;
  return __builtin_amdgcn_perm(b, a, 0x07060302); // { b[31:16], a[31:16] }
}
#endif
// Truncating pack: hot path (activations) — single v_perm_b32.
__device__ __forceinline__ unsigned int pkbf_tr(float lo, float hi) {
  return __builtin_amdgcn_perm(__float_as_uint(hi), __float_as_uint(lo), 0x07060302);
}
// Truncating scalar bf16: (bits >> 16) folds into ds_store_b16_d16_hi.
__device__ __forceinline__ unsigned short f2bf_tr(float x) {
  return (unsigned short)(__float_as_uint(x) >> 16);
}

// tanh-GELU: native v_tanh_f32 when available (1 trans op), else clamped
// rational tanh (1 rcp trans).
__device__ __forceinline__ float gelu_tanh(float x) {
  float y = x * (0.7978845608f + 0.0356774081f * x * x);
#if __has_builtin(__builtin_amdgcn_tanhf)
  float th = __builtin_amdgcn_tanhf(y);
#elif __has_builtin(__builtin_amdgcn_tanh_f32)
  float th = __builtin_amdgcn_tanh_f32(y);
#else
  float y2 = y * y;
  float th = __fdividef(y * (27.0f + y2), 27.0f + 9.0f * y2);
  th = __builtin_fminf(__builtin_fmaxf(th, -1.0f), 1.0f);
#endif
  float hx = 0.5f * x;
  return __builtin_fmaf(hx, th, hx);
}
__device__ __forceinline__ f32x8 wmma_bf16(u32x8 a, u32x8 b, f32x8 c) {
  return __builtin_amdgcn_wmma_f32_16x16x32_bf16(
      false, __builtin_bit_cast(bf16x16, a),
      false, __builtin_bit_cast(bf16x16, b),
      (short)0, c, false, false);
}
// Contiguous 8-dword LDS fragment as 2x ds_load_b128.
__device__ __forceinline__ u32x8 ld_frag(const unsigned int* p) {
  uint4 q0 = *(const uint4*)p;
  uint4 q1 = *(const uint4*)(p + 4);
  u32x8 r;
  r[0] = q0.x; r[1] = q0.y; r[2] = q0.z; r[3] = q0.w;
  r[4] = q1.x; r[5] = q1.y; r[6] = q1.z; r[7] = q1.w;
  return r;
}

__global__ void __launch_bounds__(TPB)
MLPBaseline_75617194213652_kernel(
    const float* __restrict__ nf, const int* __restrict__ eidx,
    const float* __restrict__ ef,
    const float* __restrict__ W1, const float* __restrict__ b1,
    const float* __restrict__ W2, const float* __restrict__ b2,
    const float* __restrict__ W3, const float* __restrict__ b3,
    float* __restrict__ out, int E, int numTiles)
{
  // Weights FRAG-MAJOR: dword d = (frag*32 + lane)*8 + v holds the bf16 pair
  // (k,k+1) of this lane's B-fragment dword v.
  __shared__ __align__(16) unsigned int sW1d[40 * 32 * 8];   // 40960 B (8 j x 5 ck)
  __shared__ __align__(16) unsigned int sW2d[16 * 32 * 8];   // 16384 B (4 ck x 4 j2)
  __shared__ __align__(16) unsigned short sStage[WAVES_PER_BLOCK * 512]; // 4096 B

  const int tid = threadIdx.x;
  for (int d = tid; d < 40 * 32 * 8; d += TPB) {
    int v = d & 7, l = (d >> 3) & 31, f = d >> 8;      // f = j*5 + ck
    int ck = f % 5, j = f / 5;
    int c = j * 16 + (l & 15);
    int k = ck * 32 + ((l >> 4) ? 8 : 0) + 2 * v + ((v >= 4) ? 8 : 0);
    sW1d[d] = pkbf_rnd(W1[k * H1 + c], W1[(k + 1) * H1 + c]);
  }
  for (int d = tid; d < 16 * 32 * 8; d += TPB) {
    int v = d & 7, l = (d >> 3) & 31, f = d >> 8;      // f = ck2*4 + j2
    int j2 = f & 3, ck2 = f >> 2;
    int c = j2 * 16 + (l & 15);
    int k = ck2 * 32 + ((l >> 4) ? 8 : 0) + 2 * v + ((v >= 4) ? 8 : 0);
    sW2d[d] = pkbf_rnd(W2[k * H2 + c], W2[(k + 1) * H2 + c]);
  }
  __syncthreads();

  const int lane = tid & 31;
  const int col  = lane & 15;
  const int hi   = lane >> 4;
  const int waveId = tid >> 5;
  unsigned short* st = sStage + waveId * 512;         // 16x32 bf16 staging tile
  unsigned int*  stw = (unsigned int*)st;
  const unsigned int* w1l = sW1d + lane * 8;
  const unsigned int* w2l = sW2d + lane * 8;

  // ---- loop-invariant per-lane registers: W3 fragments + biases ----
  u32x8 w3r[2];
#pragma unroll
  for (int ck3 = 0; ck3 < 2; ++ck3)
#pragma unroll
    for (int v = 0; v < 8; ++v) {
      int k = ck3 * 32 + hi * 8 + 2 * v + ((v >= 4) ? 8 : 0);
      w3r[ck3][v] = (col < NC) ? pkbf_rnd(W3[k * NC + col], W3[(k + 1) * NC + col]) : 0u;
    }
  float rb1[8], rb2[4];
#pragma unroll
  for (int j = 0; j < 8; ++j) rb1[j] = b1[j * 16 + col];
#pragma unroll
  for (int j2 = 0; j2 < 4; ++j2) rb2[j2] = b2[j2 * 16 + col];
  const float rb3 = (col < NC) ? b3[col] : 0.0f;

  const int gw = blockIdx.x * WAVES_PER_BLOCK + waveId;
  const int stride = gridDim.x * WAVES_PER_BLOCK;

  for (int tile = gw; tile < numTiles; tile += stride) {
    int e = tile * 16 + col;
    if (e >= E) e = E - 1;                    // clamp gather; stores masked later
    const int   si = eidx[e];
    const int   ti = eidx[E + e];
    const float* sp = nf + (size_t)si * D_NODE;
    const float* tp = nf + (size_t)ti * D_NODE;
    const float* ep = ef + (size_t)e  * D_EDGE;

    // ---- gather layer-1 A fragments (A layout: k = ck*32 + 2v + 8*(hi+(v>=4))) ----
    u32x8 a1[5];
#pragma unroll
    for (int ck = 0; ck < 5; ++ck) {
      const float* rp = (ck < 2) ? (sp + ck * 32)
                      : (ck < 4) ? (tp + (ck - 2) * 32)
                                 : ep;
      float4 q0 = *(const float4*)(rp + 8 * hi);
      float4 q1 = *(const float4*)(rp + 8 * hi + 4);
      float4 q2 = *(const float4*)(rp + 16 + 8 * hi);
      float4 q3 = *(const float4*)(rp + 20 + 8 * hi);
      a1[ck][0] = pkbf_tr(q0.x, q0.y); a1[ck][1] = pkbf_tr(q0.z, q0.w);
      a1[ck][2] = pkbf_tr(q1.x, q1.y); a1[ck][3] = pkbf_tr(q1.z, q1.w);
      a1[ck][4] = pkbf_tr(q2.x, q2.y); a1[ck][5] = pkbf_tr(q2.z, q2.w);
      a1[ck][6] = pkbf_tr(q3.x, q3.y); a1[ck][7] = pkbf_tr(q3.z, q3.w);
    }

    // ---- layer-2 accumulators (16x64), bias-seeded ----
    f32x8 c2[4];
#pragma unroll
    for (int j2 = 0; j2 < 4; ++j2) {
      float bv = rb2[j2];
      f32x8 tmp = {bv, bv, bv, bv, bv, bv, bv, bv};
      c2[j2] = tmp;
    }

    // ---- layer 1 (160->128), interleaved with layer-2 K chunks ----
#pragma unroll
    for (int j = 0; j < 8; ++j) {
      float bv = rb1[j];
      f32x8 c1 = {bv, bv, bv, bv, bv, bv, bv, bv};
#pragma unroll
      for (int ck = 0; ck < 5; ++ck)
        c1 = wmma_bf16(a1[ck], ld_frag(w1l + (j * 5 + ck) * 256), c1);
#pragma unroll
      for (int r = 0; r < 8; ++r)
        st[(r + 8 * hi) * 32 + 16 * (j & 1) + col] = f2bf_tr(gelu_tanh(c1[r]));
      if (j & 1) {
        int ck2 = j >> 1;
        const unsigned int* sb = stw + col * 16 + 4 * hi;
        uint4 ql = *(const uint4*)sb;
        uint4 qh = *(const uint4*)(sb + 8);
        u32x8 a2;
        a2[0] = ql.x; a2[1] = ql.y; a2[2] = ql.z; a2[3] = ql.w;
        a2[4] = qh.x; a2[5] = qh.y; a2[6] = qh.z; a2[7] = qh.w;
#pragma unroll
        for (int j2 = 0; j2 < 4; ++j2)
          c2[j2] = wmma_bf16(a2, ld_frag(w2l + (ck2 * 4 + j2) * 256), c2[j2]);
      }
    }

    // ---- layer 3 (64->8, N padded to 16), W3 fragments in registers ----
    f32x8 c3 = {rb3, rb3, rb3, rb3, rb3, rb3, rb3, rb3};
#pragma unroll
    for (int j2 = 0; j2 < 4; ++j2) {
#pragma unroll
      for (int r = 0; r < 8; ++r)
        st[(r + 8 * hi) * 32 + 16 * (j2 & 1) + col] = f2bf_tr(gelu_tanh(c2[j2][r]));
      if (j2 & 1) {
        const unsigned int* sb = stw + col * 16 + 4 * hi;
        uint4 ql = *(const uint4*)sb;
        uint4 qh = *(const uint4*)(sb + 8);
        u32x8 a3;
        a3[0] = ql.x; a3[1] = ql.y; a3[2] = ql.z; a3[3] = ql.w;
        a3[4] = qh.x; a3[5] = qh.y; a3[6] = qh.z; a3[7] = qh.w;
        c3 = wmma_bf16(a3, w3r[j2 >> 1], c3);
      }
    }

    // ---- store 16x8 output tile (f32) ----
    if (col < NC) {
#pragma unroll
      for (int r = 0; r < 8; ++r) {
        int oe = tile * 16 + r + 8 * hi;
        if (oe < E) out[(size_t)oe * NC + col] = c3[r];
      }
    }
  }
}

extern "C" void kernel_launch(void* const* d_in, const int* in_sizes, int n_in,
                              void* d_out, int out_size, void* d_ws, size_t ws_size,
                              hipStream_t stream) {
  const float* nf = (const float*)d_in[0];
  const int*   ei = (const int*)d_in[1];
  const float* ef = (const float*)d_in[2];
  const float* W1 = (const float*)d_in[3];
  const float* b1 = (const float*)d_in[4];
  const float* W2 = (const float*)d_in[5];
  const float* b2 = (const float*)d_in[6];
  const float* W3 = (const float*)d_in[7];
  const float* b3 = (const float*)d_in[8];
  float* out = (float*)d_out;

  int E = in_sizes[1] / 2;                       // edge_index is (2, E)
  if (E <= 0) return;
  int numTiles = (E + 15) / 16;                  // 16 edges per wave-tile
  int blocks = (numTiles + WAVES_PER_BLOCK - 1) / WAVES_PER_BLOCK;
  if (blocks > 2048) blocks = 2048;              // persistent waves, grid-strided

  MLPBaseline_75617194213652_kernel<<<blocks, TPB, 0, stream>>>(
      nf, ei, ef, W1, b1, W2, b2, W3, b3, out, E, numTiles);
}
